// AllSetTransformerLayer_26852135535174
// MI455X (gfx1250) — compile-verified, matrix operand order checked by reading, not run on hardware
//
#include <hip/hip_runtime.h>
#include <math.h>

typedef _Float16 half_t;
typedef __attribute__((ext_vector_type(16))) half_t v16h;
typedef __attribute__((ext_vector_type(8)))  half_t v8h;
typedef __attribute__((ext_vector_type(8)))  float  v8f;

#define NN     20000   // nodes
#define NHED   5003    // hyperedges
#define NHEDP  5008    // padded to multiple of 16 for unguarded WMMA A-loads
#define KDEG   16
#define NE     (NN*KDEG)
#define C0     256
#define HID    512
#define NHEADS 4
#define DHEAD  128
#define MAXD   96      // max hedge degree (true max is 64)

// ---------------- conversion / folding kernels ----------------
__global__ void k_conv_x(const float* __restrict__ x, half_t* __restrict__ o, int n){
  int i = blockIdx.x*256 + threadIdx.x;
  if (i < n) o[i] = (half_t)x[i];
}
// Pack f32 weights straight into the WMMA B-fragment-packed f16 layout:
//   dst[((kb*NT + nt)*32 + lane)*16 + i]  with kk = kb*32 + hi(lane) + i + (i<8?0:8),
//   n = nt*16 + (lane&15).  Each lane's 16 halves become two contiguous b128 loads.
// mode 0: W-style src[O, Kd] used as x@W.T  -> element(k,n) = src[n*Kd + k]
// mode 1: V-style src[H, Kd, 128]           -> element(k,n) = src[((n>>7)*Kd + k)*128 + (n&127)]
__global__ void k_pack_B(const float* __restrict__ src, half_t* __restrict__ dst,
                         int Kd, int N, int mode){
  int idx = blockIdx.x*256 + threadIdx.x;
  if (idx >= Kd*N) return;
  int i    = idx & 15;
  int r    = idx >> 4;
  int lane = r & 31; r >>= 5;
  int NT   = N >> 4;
  int nt   = r % NT;
  int kb   = r / NT;
  int hi   = (lane >= 16) ? 8 : 0;
  int kk   = kb*32 + hi + i + ((i < 8) ? 0 : 8);
  int n    = nt*16 + (lane & 15);
  float v = (mode == 0) ? src[(size_t)n*Kd + kk]
                        : src[((size_t)(n >> 7)*Kd + kk)*128 + (n & 127)];
  dst[idx] = (half_t)v;
}
// fold seed query into K: KQ[c,h] = sum_d K[h,c,d]*Q[h,0,d]   (QN==1)
__global__ void k_kq_fold(const float* __restrict__ Kw, const float* __restrict__ Qw,
                          float* __restrict__ KQ, int C){
  int i = blockIdx.x*256 + threadIdx.x;
  if (i >= C*NHEADS) return;
  int c = i/NHEADS, h = i - c*NHEADS;
  float s = 0.f;
  for (int d = 0; d < DHEAD; ++d)
    s += Kw[((size_t)h*C + c)*DHEAD + d] * Qw[h*DHEAD + d];
  KQ[i] = s;
}
// logits[n,h] = sum_c x[n,c] * KQ[c,h]   (x f32 or f16, one of x/xh null)
__global__ void k_logits(const float* __restrict__ x, const half_t* __restrict__ xh,
                         const float* __restrict__ KQ, float* __restrict__ out, int n, int C){
  int i = blockIdx.x*256 + threadIdx.x;
  if (i >= n*NHEADS) return;
  int r = i >> 2, h = i & 3;
  float s = 0.f;
  if (x)  { const float*  p = x  + (size_t)r*C; for (int c=0;c<C;++c) s += p[c]*KQ[c*NHEADS+h]; }
  else    { const half_t* p = xh + (size_t)r*C; for (int c=0;c<C;++c) s += (float)p[c]*KQ[c*NHEADS+h]; }
  out[i] = s;
}

// ---------------- WMMA GEMM: D = act(A[M,K] @ B[K,N] + bias), f16 in / f32 acc --------------
// block = 128 threads = 4 waves; each wave covers 16 rows x 64 cols (4 tiles, A frag reused).
// A rows must be padded to a multiple of 16 (pad rows zeroed) -> unguarded b128 loads.
// B is pre-packed into fragment layout -> two contiguous b128 loads per fragment.
__global__ __launch_bounds__(128) void k_gemm_wmma(
    const half_t* __restrict__ A, const half_t* __restrict__ Bp,
    const float* __restrict__ bias, float* __restrict__ outF,
    half_t* __restrict__ outH, int M, int N, int Kd, int relu){
  const int wave = threadIdx.x >> 5;
  const int lane = threadIdx.x & 31;
  const int l16  = lane & 15;
  const int hi   = (lane >> 4) << 3;       // K-half select per ISA 16-bit layout
  const int tileM   = blockIdx.y * 16;
  const int colBase = blockIdx.x * 256 + wave * 64;
  if (colBase >= N) return;                // wave-uniform
  const int NT  = N >> 4;
  const int nt0 = colBase >> 4;
  const half_t* Arow = A + (size_t)(tileM + l16) * Kd + hi;

  v8f accs[4] = { v8f{}, v8f{}, v8f{}, v8f{} };
  const int KB = Kd >> 5;
  for (int kb = 0; kb < KB; ++kb){
    const v8h* pa = (const v8h*)(Arow + kb*32);
    const v8h alo = pa[0], ahi = pa[2];    // k0+hi..+7 and k0+hi+16..+23
    const v16h af = __builtin_shufflevector(alo, ahi, 0,1,2,3,4,5,6,7,8,9,10,11,12,13,14,15);
    const half_t* bbase = Bp + (((size_t)kb*NT + nt0)*32 + lane)*16;
    #pragma unroll
    for (int t = 0; t < 4; ++t){
      const v8h* pb = (const v8h*)(bbase + t*(32*16));
      const v8h blo = pb[0], bhi = pb[1];
      const v16h bf = __builtin_shufflevector(blo, bhi, 0,1,2,3,4,5,6,7,8,9,10,11,12,13,14,15);
      accs[t] = __builtin_amdgcn_wmma_f32_16x16x32_f16(false, af, false, bf,
                                                       (short)0, accs[t], false, false);
    }
  }
  #pragma unroll
  for (int t = 0; t < 4; ++t){
    const int n = colBase + t*16 + l16;
    const float bv = bias ? bias[n] : 0.f;
    #pragma unroll
    for (int r = 0; r < 8; ++r){
      const int m = tileM + r + ((lane >= 16) ? 8 : 0);   // C/D VGPR layout
      if (m < M){
        float v = accs[t][r] + bv;
        if (relu) v = fmaxf(v, 0.f);
        if (outF) outF[(size_t)m*N + n] = v;
        if (outH) outH[(size_t)m*N + n] = (half_t)v;
      }
    }
  }
}

// ---------------- CSR build (targets = hyperedges) ----------------
__global__ void k_count(const int* __restrict__ hedge, int* __restrict__ cnt){
  int e = blockIdx.x*256 + threadIdx.x;
  if (e < NE) atomicAdd(&cnt[hedge[e]], 1);
}
// two-level exclusive scan over NHED counts, single block of 256 threads
__global__ void k_scan(const int* __restrict__ cnt, int* __restrict__ offs, int* __restrict__ curs){
  const int tid = threadIdx.x;
  const int CH  = 20;                       // 256*20 = 5120 >= NHED
  const int base = tid*CH;
  int s = 0;
  for (int j = 0; j < CH; ++j){ int i = base+j; if (i < NHED) s += cnt[i]; }
  __shared__ int ps[256];
  ps[tid] = s;
  __syncthreads();
  for (int o = 1; o < 256; o <<= 1){
    int v = (tid >= o) ? ps[tid-o] : 0;
    __syncthreads();
    ps[tid] += v;
    __syncthreads();
  }
  int run = ps[tid] - s;                    // exclusive prefix of this chunk
  for (int j = 0; j < CH; ++j){
    int i = base+j;
    if (i < NHED){ offs[i] = run; curs[i] = run; run += cnt[i]; }
  }
  if (tid == 255) offs[NHED] = run;
}
__global__ void k_scatter(const int* __restrict__ hedge, const int* __restrict__ node,
                          int* __restrict__ curs, int* __restrict__ csrc){
  int e = blockIdx.x*256 + threadIdx.x;
  if (e >= NE) return;
  int t = hedge[e];
  int pos = atomicAdd(&curs[t], 1);
  csrc[pos] = node[e];
}

// ---------------- segment softmax + weighted aggregation ----------------
// one block (4 waves = 4 heads) per target; CSR mode (offs!=null) or fixed-16 mode
__global__ void k_agg(const int* __restrict__ offs, const int* __restrict__ csrc,
                      const int* __restrict__ fixed_src, const half_t* __restrict__ xV,
                      const float* __restrict__ logits, const float* __restrict__ Qseed,
                      float* __restrict__ seat){
  const int t = blockIdx.x;
  __shared__ int   s_raw[MAXD];
  __shared__ int   s_src[MAXD];
  __shared__ float s_log[NHEADS][MAXD];
  int beg, deg;
  if (offs){ beg = offs[t]; deg = offs[t+1] - beg; if (deg > MAXD) deg = MAXD; }
  else     { beg = t*KDEG;  deg = KDEG; }
  const int tid = threadIdx.x;
  const int* srcp = offs ? (csrc + beg) : (fixed_src + beg);
  for (int m = tid; m < deg; m += 128) s_raw[m] = srcp[m];
  __syncthreads();
  if (offs){
    // parallel rank sort (members are unique) -> deterministic fp order
    for (int m = tid; m < deg; m += 128){
      int v = s_raw[m], r = 0;
      for (int j = 0; j < deg; ++j) r += (s_raw[j] < v);
      s_src[r] = v;
    }
  } else {
    for (int m = tid; m < deg; m += 128) s_src[m] = s_raw[m];
  }
  __syncthreads();
  const int h = tid >> 5, lane = tid & 31;
  for (int m = lane; m < deg; m += 32) s_log[h][m] = logits[s_src[m]*NHEADS + h];
  __syncthreads();
  float mx = -3.402823466e38f;
  for (int m = lane; m < deg; m += 32) mx = fmaxf(mx, s_log[h][m]);
  for (int o = 16; o > 0; o >>= 1) mx = fmaxf(mx, __shfl_xor(mx, o, 32));
  float sum = 0.f;
  for (int m = lane; m < deg; m += 32) sum += __expf(s_log[h][m] - mx);
  for (int o = 16; o > 0; o >>= 1) sum += __shfl_xor(sum, o, 32);
  const float inv = (sum > 0.f) ? (1.f/sum) : 0.f;
  float acc[4] = {0.f, 0.f, 0.f, 0.f};
  for (int m = 0; m < deg; ++m){
    const float w = __expf(s_log[h][m] - mx) * inv;
    const half_t* rowp = xV + (size_t)s_src[m]*HID + h*DHEAD;
    #pragma unroll
    for (int j = 0; j < 4; ++j) acc[j] += w * (float)rowp[lane + 32*j];
  }
  #pragma unroll
  for (int j = 0; j < 4; ++j){
    const int d = lane + 32*j;
    seat[(size_t)t*HID + h*DHEAD + d] = acc[j] + Qseed[h*DHEAD + d];
  }
}

// ---------------- fused (residual-relu) + LayerNorm (+relu) over 512 cols ----------------
__global__ void k_ln(const float* __restrict__ a, const float* __restrict__ f,
                     const float* __restrict__ g, const float* __restrict__ bt,
                     float* __restrict__ outF, half_t* __restrict__ outH, int relu_out){
  const int r = blockIdx.x, tid = threadIdx.x;
  const size_t base = (size_t)r * HID;
  float t0 = a[base + tid], t1 = a[base + tid + 256];
  if (f){ t0 += fmaxf(f[base + tid], 0.f); t1 += fmaxf(f[base + tid + 256], 0.f); }
  __shared__ float s1[256], s2[256];
  s1[tid] = t0 + t1; s2[tid] = t0*t0 + t1*t1;
  __syncthreads();
  for (int s = 128; s > 0; s >>= 1){
    if (tid < s){ s1[tid] += s1[tid+s]; s2[tid] += s2[tid+s]; }
    __syncthreads();
  }
  const float mean = s1[0] * (1.f/HID);
  const float var  = s2[0] * (1.f/HID) - mean*mean;
  const float rs   = rsqrtf(var + 1e-5f);
  float y0 = (t0 - mean)*rs*g[tid]     + bt[tid];
  float y1 = (t1 - mean)*rs*g[tid+256] + bt[tid+256];
  if (relu_out){ y0 = fmaxf(y0, 0.f); y1 = fmaxf(y1, 0.f); }
  if (outF){ outF[base+tid] = y0; outF[base+tid+256] = y1; }
  if (outH){ outH[base+tid] = (half_t)y0; outH[base+tid+256] = (half_t)y1; }
}

// ---------------- host-side orchestration ----------------
extern "C" void kernel_launch(void* const* d_in, const int* in_sizes, int n_in,
                              void* d_out, int out_size, void* d_ws, size_t ws_size,
                              hipStream_t stream) {
  (void)in_sizes; (void)n_in; (void)out_size; (void)ws_size;
  const float* x0       = (const float*)d_in[0];
  const int*   node_idx = (const int*)  d_in[1];
  const int*   hedge_idx= (const int*)  d_in[2];
  const float* ve_K = (const float*)d_in[4],  *ve_Q = (const float*)d_in[5],  *ve_V = (const float*)d_in[6];
  const float* ve_w1= (const float*)d_in[7],  *ve_b1= (const float*)d_in[8];
  const float* ve_w2= (const float*)d_in[9],  *ve_b2= (const float*)d_in[10];
  const float* ve_l0g=(const float*)d_in[11], *ve_l0b=(const float*)d_in[12];
  const float* ve_l1g=(const float*)d_in[13], *ve_l1b=(const float*)d_in[14];
  const float* ev_K = (const float*)d_in[15], *ev_Q = (const float*)d_in[16], *ev_V = (const float*)d_in[17];
  const float* ev_w1= (const float*)d_in[18], *ev_b1= (const float*)d_in[19];
  const float* ev_w2= (const float*)d_in[20], *ev_b2= (const float*)d_in[21];
  const float* ev_l0g=(const float*)d_in[22], *ev_l0b=(const float*)d_in[23];
  const float* ev_l1g=(const float*)d_in[24], *ev_l1b=(const float*)d_in[25];
  float* out = (float*)d_out;

  // ---- carve workspace (~200 MB) ----
  char* ws = (char*)d_ws; size_t off = 0;
  auto carve = [&](size_t bytes)->char* {
    char* p = ws + off; off = (off + bytes + 255) & ~(size_t)255; return p;
  };
  half_t* whV1  = (half_t*)carve((size_t)C0 *HID*2);   // packed B layouts
  half_t* whW11 = (half_t*)carve((size_t)HID*HID*2);
  half_t* whW12 = (half_t*)carve((size_t)HID*HID*2);
  half_t* whV2  = (half_t*)carve((size_t)HID*HID*2);
  half_t* whW21 = (half_t*)carve((size_t)HID*HID*2);
  half_t* whW22 = (half_t*)carve((size_t)HID*HID*2);
  float*  kq1   = (float*) carve((size_t)C0 *NHEADS*4);
  float*  kq2   = (float*) carve((size_t)HID*NHEADS*4);
  half_t* xh0   = (half_t*)carve((size_t)NN*C0*2);
  float*  log1  = (float*) carve((size_t)NN*NHEADS*4);
  half_t* xv1   = (half_t*)carve((size_t)NN*HID*2);
  int*    cnt   = (int*)   carve((size_t)NHED*4);
  int*    offs  = (int*)   carve((size_t)(NHED+1)*4);
  int*    curs  = (int*)   carve((size_t)NHED*4);
  int*    csrc  = (int*)   carve((size_t)NE*4);
  float*  seat1 = (float*) carve((size_t)NHED*HID*4);
  half_t* ln01h = (half_t*)carve((size_t)NHEDP*HID*2); // padded A buffers
  half_t* hb1   = (half_t*)carve((size_t)NHEDP*HID*2);
  float*  ff1   = (float*) carve((size_t)NHED*HID*4);
  half_t* x1h   = (half_t*)carve((size_t)NHEDP*HID*2);
  float*  log2  = (float*) carve((size_t)NHED*NHEADS*4);
  half_t* xv2   = (half_t*)carve((size_t)NHED*HID*2);
  float*  seat2 = (float*) carve((size_t)NN*HID*4);
  half_t* ln02h = (half_t*)carve((size_t)NN*HID*2);
  half_t* hb2   = (half_t*)carve((size_t)NN*HID*2);
  float*  ff2   = (float*) carve((size_t)NN*HID*4);

  // zero pad rows of M=5003 A-side buffers (rows NHED..NHEDP-1)
  const size_t padB = (size_t)(NHEDP - NHED)*HID*2;
  hipMemsetAsync(ln01h + (size_t)NHED*HID, 0, padB, stream);
  hipMemsetAsync(hb1   + (size_t)NHED*HID, 0, padB, stream);
  hipMemsetAsync(x1h   + (size_t)NHED*HID, 0, padB, stream);
  hipMemsetAsync(cnt, 0, (size_t)NHED*4, stream);

  // ---- weight staging (packed f16) + query folding ----
  k_pack_B<<<(C0*HID+255)/256, 256, 0, stream>>>(ve_V, whV1, C0,  HID, 1);
  k_pack_B<<<(HID*HID+255)/256, 256, 0, stream>>>(ev_V, whV2, HID, HID, 1);
  k_pack_B<<<(HID*HID+255)/256, 256, 0, stream>>>(ve_w1, whW11, HID, HID, 0);
  k_pack_B<<<(HID*HID+255)/256, 256, 0, stream>>>(ve_w2, whW12, HID, HID, 0);
  k_pack_B<<<(HID*HID+255)/256, 256, 0, stream>>>(ev_w1, whW21, HID, HID, 0);
  k_pack_B<<<(HID*HID+255)/256, 256, 0, stream>>>(ev_w2, whW22, HID, HID, 0);
  k_kq_fold<<<(C0*NHEADS+255)/256, 256, 0, stream>>>(ve_K, ve_Q, kq1, C0);
  k_kq_fold<<<(HID*NHEADS+255)/256, 256, 0, stream>>>(ev_K, ev_Q, kq2, HID);
  k_conv_x<<<((NN*C0)+255)/256, 256, 0, stream>>>(x0, xh0, NN*C0);

  // ---- block 1: vertex -> hyperedge ----
  k_logits<<<(NN*NHEADS+255)/256, 256, 0, stream>>>(x0, nullptr, kq1, log1, NN, C0);
  k_gemm_wmma<<<dim3(HID/256, NN/16), 128, 0, stream>>>(
      xh0, whV1, nullptr, nullptr, xv1, NN, HID, C0, 0);
  k_count  <<<(NE+255)/256, 256, 0, stream>>>(hedge_idx, cnt);
  k_scan   <<<1, 256, 0, stream>>>(cnt, offs, curs);
  k_scatter<<<(NE+255)/256, 256, 0, stream>>>(hedge_idx, node_idx, curs, csrc);
  k_agg<<<NHED, 128, 0, stream>>>(offs, csrc, nullptr, xv1, log1, ve_Q, seat1);
  k_ln <<<NHED, 256, 0, stream>>>(seat1, nullptr, ve_l0g, ve_l0b, seat1, ln01h, 0);
  k_gemm_wmma<<<dim3(HID/256, NHEDP/16), 128, 0, stream>>>(
      ln01h, whW11, ve_b1, nullptr, hb1, NHED, HID, HID, 1);
  k_gemm_wmma<<<dim3(HID/256, NHEDP/16), 128, 0, stream>>>(
      hb1, whW12, ve_b2, ff1, nullptr, NHED, HID, HID, 0);
  k_ln <<<NHED, 256, 0, stream>>>(seat1, ff1, ve_l1g, ve_l1b, nullptr, x1h, 1);

  // ---- block 2: hyperedge -> vertex ----
  k_logits<<<(NHED*NHEADS+255)/256, 256, 0, stream>>>(nullptr, x1h, kq2, log2, NHED, HID);
  k_gemm_wmma<<<dim3(HID/256, NHEDP/16), 128, 0, stream>>>(
      x1h, whV2, nullptr, nullptr, xv2, NHED, HID, HID, 0);
  k_agg<<<NN, 128, 0, stream>>>(nullptr, nullptr, hedge_idx, xv2, log2, ev_Q, seat2);
  k_ln <<<NN, 256, 0, stream>>>(seat2, nullptr, ev_l0g, ev_l0b, seat2, ln02h, 0);
  k_gemm_wmma<<<dim3(HID/256, NN/16), 128, 0, stream>>>(
      ln02h, whW21, ev_b1, nullptr, hb2, NN, HID, HID, 1);
  k_gemm_wmma<<<dim3(HID/256, NN/16), 128, 0, stream>>>(
      hb2, whW22, ev_b2, ff2, nullptr, NN, HID, HID, 0);
  k_ln <<<NN, 256, 0, stream>>>(seat2, ff2, ev_l1g, ev_l1b, out, nullptr, 1);
}
